// LSTM_A_47751446397532
// MI455X (gfx1250) — matrix operand order repaired and checked by reading, not compiled, not run on hardware
//
#include <hip/hip_runtime.h>

// ---------------- problem constants ----------------
#define Bb    128
#define Tt    800
#define STRKn 3
#define Ll    64        // sent_max_len
#define Aa    60        // alphabet
#define Hh    400       // lstm hidden
#define KMIX  10
#define G4H   1600      // 4*H
// ---------------- decomposition -------------------
#define NWG   10        // workgroups in cluster
#define COLS  160       // gate columns per WG (10 N-tiles)
#define NT    10
#define KTOT  480       // padded K: 64 (x|w) + 416 (h padded)
#define KH    416       // padded hidden K
#define HCOLS 40        // hidden columns per WG for c/h update

typedef __attribute__((ext_vector_type(16))) _Float16 v16h;
typedef __attribute__((ext_vector_type(8)))  float    v8f;
typedef __attribute__((ext_vector_type(4)))  unsigned u32x4;

union Frag { v16h v; u32x4 q[2]; };

__device__ __forceinline__ float sigm(float x) { return 1.0f / (1.0f + __expf(-x)); }

// ---------------------------------------------------------------
// Prep: fuse/convert weights to f16 with K padding, fuse biases,
// convert h0 to f16 state buffer. Deterministic, graph-safe.
// ---------------------------------------------------------------
__global__ void lstm_attn_prep(const float* __restrict__ Wih, const float* __restrict__ Whh,
                               const float* __restrict__ bih, const float* __restrict__ bhh,
                               const float* __restrict__ Wp,  const float* __restrict__ h0,
                               _Float16* __restrict__ W16, _Float16* __restrict__ Wp16,
                               _Float16* __restrict__ h16, float* __restrict__ biasg) {
  int i = blockIdx.x * blockDim.x + threadIdx.x;
  const int NW = G4H * KTOT;
  if (i < NW) {                               // combined [Wih | pad | Whh | pad] rows
    int n = i / KTOT, k = i % KTOT;
    float v = 0.f;
    if (k < STRKn + Aa)            v = Wih[n * (STRKn + Aa) + k];
    else if (k >= 64 && k < 64+Hh) v = Whh[n * Hh + (k - 64)];
    W16[i] = (_Float16)v;
    return;
  }
  i -= NW;
  if (i < 32 * KH) {                          // Wp padded to [32][416]
    int j = i / KH, k = i % KH;
    Wp16[i] = (_Float16)((j < 3*KMIX && k < Hh) ? Wp[j * Hh + k] : 0.f);
    return;
  }
  i -= 32 * KH;
  if (i < Bb * KH) {                          // h0 -> f16 padded [128][416]
    int b = i / KH, k = i % KH;
    h16[i] = (_Float16)(k < Hh ? h0[b * Hh + k] : 0.f);
    return;
  }
  i -= Bb * KH;
  if (i < G4H) biasg[i] = bih[i] + bhh[i];
}

// ---------------------------------------------------------------
// Persistent cluster kernel: 10 workgroups x 256 threads (8 waves)
// 2 cluster barriers per timestep.
// ---------------------------------------------------------------
__global__ __launch_bounds__(256, 1)
void lstm_attn_main(const float* __restrict__ strks, const float* __restrict__ onehots,
                    const float* __restrict__ sents_m, const float* __restrict__ w_prev,
                    const float* __restrict__ k_prev, const float* __restrict__ c0,
                    const float* __restrict__ bp,
                    const _Float16* __restrict__ W16, const _Float16* __restrict__ Wp16,
                    _Float16* __restrict__ h16, const float* __restrict__ biasg,
                    float* __restrict__ gates,
                    float* __restrict__ out_hid1, float* __restrict__ out_hf,
                    float* __restrict__ out_cf,   float* __restrict__ out_win,
                    float* __restrict__ out_wf,   float* __restrict__ out_kf,
                    float* __restrict__ out_phil) {
  extern __shared__ char smem[];
  _Float16* W_s    = (_Float16*)(smem);               // [160][480] f16  153600 B
  _Float16* Wp_s   = (_Float16*)(smem + 153600);      // [32][416]  f16   26624 B
  _Float16* x_s    = (_Float16*)(smem + 180224);      // [128][64]  f16   16384 B
  float*    c_s    = (float*)(smem + 196608);         // [128][40]        20480 B
  float*    k_s    = (float*)(smem + 217088);         // [128][10]         5120 B
  float*    abk_s  = (float*)(smem + 222208);         // [128][32]        16384 B
  float*    phi_s  = (float*)(smem + 238592);         // [128][66]        33792 B
  float*    scale_s= (float*)(smem + 272384);         // [128]              512 B
  float*    bias_s = (float*)(smem + 272896);         // [160]              640 B

  const int tid  = threadIdx.x;
  const int wg   = blockIdx.x;
  const int lane = tid & 31;
  const int wave = tid >> 5;          // M-tile id (8 waves = 128 batch rows)
  const int hlf  = lane >> 4;         // lane half per ISA fragment layout
  const int l16  = lane & 15;

  // ---------------- init ----------------
  {
    const u32x4* src = (const u32x4*)(W16 + (size_t)wg * COLS * KTOT);
    u32x4* dst = (u32x4*)W_s;
    for (int i = tid; i < COLS * KTOT / 8; i += 256) dst[i] = src[i];
    const u32x4* src2 = (const u32x4*)Wp16;
    u32x4* dst2 = (u32x4*)Wp_s;
    for (int i = tid; i < 32 * KH / 8; i += 256) dst2[i] = src2[i];
    for (int i = tid; i < COLS; i += 256) bias_s[i] = biasg[wg * COLS + i];
    for (int i = tid; i < Bb * HCOLS; i += 256) {
      int b = i / HCOLS, j = i % HCOLS;
      c_s[i] = c0[b * Hh + wg * HCOLS + j];
    }
    for (int i = tid; i < Bb * KMIX; i += 256) k_s[i] = k_prev[i];
    for (int b = tid; b < Bb; b += 256) {
      float s = 0.f;
      for (int u = 0; u < Ll; ++u) s += sents_m[b * Ll + u];
      scale_s[b] = (float)Ll / s;
      x_s[b*64 + 0]  = (_Float16)strks[((size_t)b * Tt + 0) * STRKn + 0];
      x_s[b*64 + 1]  = (_Float16)strks[((size_t)b * Tt + 0) * STRKn + 1];
      x_s[b*64 + 2]  = (_Float16)strks[((size_t)b * Tt + 0) * STRKn + 2];
      x_s[b*64 + 63] = (_Float16)0.f;
      for (int a = 0; a < Aa; ++a) x_s[b*64 + 3 + a] = (_Float16)w_prev[b * Aa + a];
    }
  }
  __syncthreads();

  const v8f vzero = {0.f,0.f,0.f,0.f,0.f,0.f,0.f,0.f};

  for (int t = 0; t < Tt; ++t) {
    // ================= Phase A: gates = [x|w|h] @ Wslice^T + bias =================
    {
      v8f acc[NT];
#pragma unroll
      for (int n = 0; n < NT; ++n) acc[n] = vzero;

      for (int kt = 0; kt < KTOT / 32; ++kt) {
        Frag fa;
        const int k0 = kt * 32 + hlf * 8;   // A-frag: 8 halves @k0, 8 @k0+16 (ISA 7.12.2)
        if (kt < 2) {
          const _Float16* asrc = &x_s[(wave * 16 + l16) * 64 + k0];
          fa.q[0] = *(const u32x4*)asrc;
          fa.q[1] = *(const u32x4*)(asrc + 16);
        } else {
          const _Float16* asrc = &h16[(size_t)(wave * 16 + l16) * KH + (k0 - 64)];
          fa.q[0] = *(const u32x4*)asrc;
          fa.q[1] = *(const u32x4*)(asrc + 16);
        }
#pragma unroll
        for (int n = 0; n < NT; ++n) {
          // B-frag: 8 halves @ half*8, 8 @ half*8+16 of weight row (N on lanes)
          Frag fbf; { const _Float16* bsrc = &W_s[(n * 16 + l16) * KTOT + kt * 32 + hlf * 8]; fbf.q[0] = *(const u32x4*)bsrc; fbf.q[1] = *(const u32x4*)(bsrc + 16); } v16h fb = fbf.v;
          acc[n] = __builtin_amdgcn_wmma_f32_16x16x32_f16(false, fa.v, false, fb,
                                                          (short)0, acc[n], false, false);
        }
      }
#pragma unroll
      for (int n = 0; n < NT; ++n) {
        const int ncl = n * 16 + l16;
        const float bv = bias_s[ncl];
        const int ncg = wg * COLS + ncl;
        const int mbase = wave * 16 + hlf * 8;   // D: VGPR r -> row r (+8 for hi lanes)
#pragma unroll
        for (int r = 0; r < 8; ++r)
          gates[(size_t)(mbase + r) * G4H + ncg] = acc[n][r] + bv;
      }
    }
    __syncthreads();
    __builtin_amdgcn_fence(__ATOMIC_RELEASE, "agent");
    __builtin_amdgcn_s_cluster_barrier();
    __builtin_amdgcn_fence(__ATOMIC_ACQUIRE, "agent");

    // ================= Phase B: LSTM cell update on owned h-columns =================
    for (int i = tid; i < Bb * HCOLS; i += 256) {
      const int b = i / HCOLS, j = i % HCOLS;
      const int jg = wg * HCOLS + j;
      const float* gr = gates + (size_t)b * G4H + jg;
      const float iv = sigm(gr[0]);
      const float fv = sigm(gr[Hh]);
      const float gv = tanhf(gr[2 * Hh]);
      const float ov = sigm(gr[3 * Hh]);
      const float c = fv * c_s[i] + iv * gv;
      c_s[i] = c;
      const float h = ov * tanhf(c);
      out_hid1[((size_t)b * Tt + t) * Hh + jg] = h;
      h16[(size_t)b * KH + jg] = (_Float16)h;
      if (t == Tt - 1) { out_hf[b * Hh + jg] = h; out_cf[b * Hh + jg] = c; }
    }
    __syncthreads();
    __builtin_amdgcn_fence(__ATOMIC_RELEASE, "agent");
    __builtin_amdgcn_s_cluster_barrier();
    __builtin_amdgcn_fence(__ATOMIC_ACQUIRE, "agent");

    // ================= Phase C: attention (redundant per WG, avoids 3rd barrier) ====
    {
      v8f acc[2] = {vzero, vzero};
      for (int kt = 0; kt < KH / 32; ++kt) {
        Frag fa;
        const int k0 = kt * 32 + hlf * 8;
        const _Float16* asrc = &h16[(size_t)(wave * 16 + l16) * KH + k0];
        fa.q[0] = *(const u32x4*)asrc;
        fa.q[1] = *(const u32x4*)(asrc + 16);
#pragma unroll
        for (int n = 0; n < 2; ++n) {
          Frag fbf; { const _Float16* bsrc = &Wp_s[(n * 16 + l16) * KH + kt * 32 + hlf * 8]; fbf.q[0] = *(const u32x4*)bsrc; fbf.q[1] = *(const u32x4*)(bsrc + 16); } v16h fb = fbf.v;
          acc[n] = __builtin_amdgcn_wmma_f32_16x16x32_f16(false, fa.v, false, fb,
                                                          (short)0, acc[n], false, false);
        }
      }
#pragma unroll
      for (int n = 0; n < 2; ++n) {
        const int col = n * 16 + l16;
        const float bv = (col < 3 * KMIX) ? bp[col] : 0.f;
        const int mbase = wave * 16 + hlf * 8;
#pragma unroll
        for (int r = 0; r < 8; ++r)
          abk_s[(mbase + r) * 32 + col] = acc[n][r] + bv;
      }
    }
    __syncthreads();
    // a=exp, b=exp, k += exp(k)
    for (int i = tid; i < Bb * KMIX; i += 256) {
      const int b = i / KMIX, j = i % KMIX;
      const float av = __expf(abk_s[b * 32 + j]);
      const float bv = __expf(abk_s[b * 32 + KMIX + j]);
      const float kk = k_s[i] + __expf(abk_s[b * 32 + 2 * KMIX + j]);
      k_s[i] = kk;
      abk_s[b * 32 + j] = av;
      abk_s[b * 32 + KMIX + j] = bv;
      abk_s[b * 32 + 2 * KMIX + j] = kk;
      if (t == Tt - 1 && wg == 0) out_kf[i] = kk;
    }
    __syncthreads();
    // phi[b][u] = scale * sum_j a*exp(-b*(k-u)^2)
    for (int i = tid; i < Bb * (Ll + 1); i += 256) {
      const int b = i / (Ll + 1), u = i % (Ll + 1);
      const float uu = (float)u;
      float s = 0.f;
#pragma unroll
      for (int j = 0; j < KMIX; ++j) {
        const float d = abk_s[b * 32 + 2 * KMIX + j] - uu;
        s += abk_s[b * 32 + j] * __expf(-abk_s[b * 32 + KMIX + j] * d * d);
      }
      s *= scale_s[b];
      phi_s[b * 66 + u] = s;
      if (t == Tt - 1 && wg == 0) out_phil[b * (Ll + 1) + u] = s;
    }
    __syncthreads();
    // w_t[b][a] = sum_u phi * onehot -> feeds x_s for next step
    for (int i = tid; i < Bb * Aa; i += 256) {
      const int b = i / Aa, a = i % Aa;
      float s = 0.f;
      for (int u = 0; u < Ll; ++u)
        s += phi_s[b * 66 + u] * onehots[((size_t)b * Ll + u) * Aa + a];
      x_s[b * 64 + 3 + a] = (_Float16)s;
      if (wg == 0) out_win[((size_t)b * Tt + t) * Aa + a] = s;
      if (t == Tt - 1 && wg == 0) out_wf[i] = s;
    }
    if (t + 1 < Tt) {
      for (int b = tid; b < Bb; b += 256) {
        x_s[b*64 + 0] = (_Float16)strks[((size_t)b * Tt + t + 1) * STRKn + 0];
        x_s[b*64 + 1] = (_Float16)strks[((size_t)b * Tt + t + 1) * STRKn + 1];
        x_s[b*64 + 2] = (_Float16)strks[((size_t)b * Tt + t + 1) * STRKn + 2];
      }
    }
    __syncthreads();  // x_s ready before next Phase A
  }
}

extern "C" void kernel_launch(void* const* d_in, const int* in_sizes, int n_in,
                              void* d_out, int out_size, void* d_ws, size_t ws_size,
                              hipStream_t stream) {
  const float* strks   = (const float*)d_in[0];
  const float* onehots = (const float*)d_in[1];
  const float* sents_m = (const float*)d_in[2];
  const float* w_prev  = (const float*)d_in[3];
  const float* k_prev  = (const float*)d_in[4];
  const float* h0      = (const float*)d_in[5];
  const float* c0      = (const float*)d_in[6];
  const float* Wih     = (const float*)d_in[7];
  const float* bih     = (const float*)d_in[8];
  const float* Whh     = (const float*)d_in[9];
  const float* bhh     = (const float*)d_in[10];
  const float* Wp      = (const float*)d_in[11];
  const float* bp      = (const float*)d_in[12];

  char* ws = (char*)d_ws;
  float*    gates = (float*)(ws + 0);            // 128*1600*4 = 819200
  _Float16* h16   = (_Float16*)(ws + 819200);    // 128*416*2  = 106496
  _Float16* W16   = (_Float16*)(ws + 925696);    // 1600*480*2 = 1536000
  _Float16* Wp16  = (_Float16*)(ws + 2461696);   // 32*416*2   = 26624
  float*    biasg = (float*)(ws + 2488320);      // 1600*4     = 6400

  float* out = (float*)d_out;
  float* out_hid1 = out;               // [128,800,400]
  float* out_hf   = out + 40960000;    // [128,400]
  float* out_cf   = out + 41011200;    // [128,400]
  float* out_win  = out + 41062400;    // [128,800,60]
  float* out_wf   = out + 47206400;    // [128,60]
  float* out_kf   = out + 47214080;    // [128,10]
  float* out_phil = out + 47215360;    // [128,65]

  const int prep_n = G4H * KTOT + 32 * KH + Bb * KH + G4H;
  lstm_attn_prep<<<(prep_n + 255) / 256, 256, 0, stream>>>(
      Wih, Whh, bih, bhh, Wp, h0, W16, Wp16, h16, biasg);

  const size_t lds_bytes = 273536;     // < 320 KB per WGP
  lstm_attn_main<<<NWG, 256, lds_bytes, stream>>>(
      strks, onehots, sents_m, w_prev, k_prev, c0, bp,
      W16, Wp16, h16, biasg, gates,
      out_hid1, out_hf, out_cf, out_win, out_wf, out_kf, out_phil);
}
